// MyRNN_84920093376826
// MI455X (gfx1250) — matrix operand order repaired and checked by reading, not compiled, not run on hardware
//
#include <hip/hip_runtime.h>
#include <math.h>
#include <stdint.h>

// ---------------------------------------------------------------------------
// Problem constants
// ---------------------------------------------------------------------------
#define T_STEPS   1024
#define DIM       300
#define H1        1024
#define H2DIM     512
#define K1        (DIM + H1)     // 1324
#define K2        (H1 + H2DIM)   // 1536
#define KD        512            // GEMM reduction dim
#define NG        50000
#define NS        25000

#define RNN_BLOCKS  32
#define RNN_THREADS 256
#define RNN_WAVES   (RNN_BLOCKS * (RNN_THREADS / 32))   // 256 waves

// ---------------------------------------------------------------------------
// CDNA5 types
// ---------------------------------------------------------------------------
typedef __attribute__((ext_vector_type(16))) __bf16 v16bf;
typedef __attribute__((ext_vector_type(8)))  float  v8f;
typedef __attribute__((ext_vector_type(4)))  unsigned int v4u;
typedef __attribute__((ext_vector_type(8)))  int    v8i32;
typedef __attribute__((ext_vector_type(4)))  int    v4i32;

// AS-qualified vector pointers for the async global->LDS builtins
typedef __attribute__((address_space(1))) v4i32 gv4i_t;  // global
typedef __attribute__((address_space(3))) v4i32 lv4i_t;  // LDS

union FragU { v16bf v; unsigned int u[8]; };

// pack two f32 into one dword of bf16 (RNE) — pure VALU op, schedulable
__device__ __forceinline__ unsigned int pack_bf16(float a, float b) {
    unsigned int r;
    asm("v_cvt_pk_bf16_f32 %0, %1, %2" : "=v"(r) : "v"(a), "v"(b));
    return r;
}

// ---------------------------------------------------------------------------
// Init kernel: zero RNN state + grid-barrier counters (run once per launch)
// ---------------------------------------------------------------------------
__global__ void init_ws_kernel(float* h1buf, float* zbuf, int* bar) {
    int i = blockIdx.x * blockDim.x + threadIdx.x;
    if (i < 2 * H1) h1buf[i] = 0.0f;
    if (i < H2DIM)  zbuf[i]  = 0.0f;
    if (i < 4)      bar[i]   = 0;
}

// ---------------------------------------------------------------------------
// Device-wide barrier (generation counter). RNN_BLOCKS workgroups resident.
// ---------------------------------------------------------------------------
__device__ __forceinline__ void grid_barrier(int* bar) {
    __syncthreads();
    __threadfence();
    if (threadIdx.x == 0) {
        int* cnt = bar;
        int* gen = bar + 1;
        int g = __hip_atomic_load(gen, __ATOMIC_ACQUIRE, __HIP_MEMORY_SCOPE_AGENT);
        int prev = __hip_atomic_fetch_add(cnt, 1, __ATOMIC_ACQ_REL, __HIP_MEMORY_SCOPE_AGENT);
        if (prev == RNN_BLOCKS - 1) {
            __hip_atomic_store(cnt, 0, __ATOMIC_RELAXED, __HIP_MEMORY_SCOPE_AGENT);
            __hip_atomic_fetch_add(gen, 1, __ATOMIC_ACQ_REL, __HIP_MEMORY_SCOPE_AGENT);
        } else {
            while (__hip_atomic_load(gen, __ATOMIC_ACQUIRE, __HIP_MEMORY_SCOPE_AGENT) == g)
                __builtin_amdgcn_s_sleep(8);
        }
    }
    __syncthreads();
}

// ---------------------------------------------------------------------------
// Sequential RNN: persistent grid, one wave per output row, lanes split K.
// W1/W2 (8.6 MB) stay L2-resident across all 1024 steps.
// ---------------------------------------------------------------------------
__global__ void __launch_bounds__(RNN_THREADS)
rnn_kernel(const float* __restrict__ X,
           const float* __restrict__ W1, const float* __restrict__ b1,
           const float* __restrict__ W2, const float* __restrict__ b2,
           const int*   __restrict__ idx,
           float* __restrict__ h2s,
           float* __restrict__ h1buf,
           const float* __restrict__ zbuf,
           int* bar) {
    const int lane = threadIdx.x & 31;
    const int wave = (blockIdx.x << 3) + (threadIdx.x >> 5);  // 0..255

    for (int t = 0; t < T_STEPS; ++t) {
        // prefetch next gathered embedding row (random access -> hide latency)
        if (threadIdx.x == 0 && t + 1 < T_STEPS)
            __builtin_prefetch(X + (size_t)idx[t + 1] * DIM, 0, 0);

        const float* e   = X + (size_t)idx[t] * DIM;
        const float* h1o = h1buf + (size_t)(t & 1) * H1;
        float*       h1n = h1buf + (size_t)((t & 1) ^ 1) * H1;
        const float* h2o = (t == 0) ? zbuf : (h2s + (size_t)(t - 1) * H2DIM);
        float*       h2n = h2s + (size_t)t * H2DIM;

        // ---- phase 1: h1n = leaky_relu(W1 @ [e; h1] + b1) ----
        #pragma unroll
        for (int rep = 0; rep < H1 / RNN_WAVES; ++rep) {
            const int o = wave + rep * RNN_WAVES;
            const float* w  = W1 + (size_t)o * K1;
            const float* wh = w + DIM;
            float acc = 0.0f;
            for (int k = lane; k < DIM; k += 32) acc += w[k]  * e[k];
            for (int k = lane; k < H1;  k += 32) acc += wh[k] * h1o[k];
            #pragma unroll
            for (int off = 16; off; off >>= 1) acc += __shfl_down(acc, off, 32);
            if (lane == 0) {
                float v = acc + b1[o];
                h1n[o] = v > 0.0f ? v : 0.01f * v;
            }
        }
        grid_barrier(bar);

        // ---- phase 2: h2n = leaky_relu(W2 @ [h1n; h2] + b2) ----
        #pragma unroll
        for (int rep = 0; rep < H2DIM / RNN_WAVES; ++rep) {
            const int o = wave + rep * RNN_WAVES;
            const float* w  = W2 + (size_t)o * K2;
            const float* wh = w + H1;
            float acc = 0.0f;
            for (int k = lane; k < H1;    k += 32) acc += w[k]  * h1n[k];
            for (int k = lane; k < H2DIM; k += 32) acc += wh[k] * h2o[k];
            #pragma unroll
            for (int off = 16; off; off >>= 1) acc += __shfl_down(acc, off, 32);
            if (lane == 0) {
                float v = acc + b2[o];
                h2n[o] = v > 0.0f ? v : 0.01f * v;
            }
        }
        grid_barrier(bar);
    }
}

// ---------------------------------------------------------------------------
// Fragment builders: batch 4 x ds_load_b128 per fragment, then 8 packs.
// Pitch 36 dwords (144B rows): 16B-aligned b128 accesses, bank-conflict-free
// (start banks 36*i mod 64 tile the 64 banks in steps of 4).
//   A 16x32 (ISA 7.12.2): lane = M%16; half selects K+0/+8 per 16-group.
//   B 32x16: lane = N%16; half selects K+0/+8 per 16-group (same as A).
// ---------------------------------------------------------------------------
#define FPITCH 36

__device__ __forceinline__ v16bf load_a_frag(const float (*Af)[FPITCH],
                                             int m_base, int lane) {
    const int m  = m_base + (lane & 15);
    const int h8 = (lane >> 4) << 3;
    const float4 q0 = *(const float4*)&Af[m][h8];
    const float4 q1 = *(const float4*)&Af[m][h8 + 4];
    const float4 q2 = *(const float4*)&Af[m][16 + h8];
    const float4 q3 = *(const float4*)&Af[m][16 + h8 + 4];
    FragU f;
    f.u[0] = pack_bf16(q0.x, q0.y); f.u[1] = pack_bf16(q0.z, q0.w);
    f.u[2] = pack_bf16(q1.x, q1.y); f.u[3] = pack_bf16(q1.z, q1.w);
    f.u[4] = pack_bf16(q2.x, q2.y); f.u[5] = pack_bf16(q2.z, q2.w);
    f.u[6] = pack_bf16(q3.x, q3.y); f.u[7] = pack_bf16(q3.z, q3.w);
    return f.v;
}

__device__ __forceinline__ v16bf load_b_frag(const float (*Bf)[FPITCH],
                                             int n_base, int lane) {
    const int n  = n_base + (lane & 15);
    const int kb = (lane >> 4) << 3;
    const float4 q0 = *(const float4*)&Bf[n][kb];
    const float4 q1 = *(const float4*)&Bf[n][kb + 4];
    const float4 q2 = *(const float4*)&Bf[n][16 + kb];
    const float4 q3 = *(const float4*)&Bf[n][16 + kb + 4];
    FragU f;
    f.u[0] = pack_bf16(q0.x, q0.y); f.u[1] = pack_bf16(q0.z, q0.w);
    f.u[2] = pack_bf16(q1.x, q1.y); f.u[3] = pack_bf16(q1.z, q1.w);
    f.u[4] = pack_bf16(q2.x, q2.y); f.u[5] = pack_bf16(q2.z, q2.w);
    f.u[6] = pack_bf16(q3.x, q3.y); f.u[7] = pack_bf16(q3.z, q3.w);
    return f.v;
}

// ---------------------------------------------------------------------------
// logits[m][n] = dot(h2s[m,:], Wrow[n,:]) + bias[n]
// 128x128 tile / workgroup, BK=32, double-buffered software pipeline:
//   B tile k+1 streamed by the TDM (tensor_load_to_lds, wave 0, TENSORcnt)
//   A tile k+1 via async global->LDS b128 (ASYNCcnt)
// while XDL runs the 8 bf16 WMMAs of chunk k. Counters used for overlap:
//   s_wait_asynccnt 4  -> previous A chunk drained, prefetch in flight
//   s_wait_tensorcnt 1 -> previous B chunk drained, prefetch in flight
// ---------------------------------------------------------------------------
__global__ void __launch_bounds__(256)
gemm_bias_kernel(const float* __restrict__ A,    // [M, KD]
                 const float* __restrict__ Wt,   // [N, KD] row-major
                 const float* __restrict__ bias, // [N]
                 float* __restrict__ out,        // [M, N]
                 int N) {
    __shared__ __align__(16) float Af[2][128][FPITCH];
    __shared__ __align__(16) float Bf[2][128][FPITCH];

    const int tid    = threadIdx.x;
    const int lane   = tid & 31;
    const int w      = tid >> 5;
    const int m_wave = (w & 3) << 5;   // 0,32,64,96
    const int n_wave = (w >> 2) << 6;  // 0,64
    const int bm     = blockIdx.y << 7;
    const int bn     = blockIdx.x << 7;

    v8f acc[2][4];
    const v8f vzero = {0.f, 0.f, 0.f, 0.f, 0.f, 0.f, 0.f, 0.f};
    #pragma unroll
    for (int i = 0; i < 2; ++i)
        #pragma unroll
        for (int j = 0; j < 4; ++j) acc[i][j] = vzero;

    const unsigned int ldsB[2] = {
        (unsigned int)(uintptr_t)&Bf[0][0][0],
        (unsigned int)(uintptr_t)&Bf[1][0][0]
    };
    const unsigned int rows_rem = (unsigned int)(N - bn);  // OOB rows zero-fill

    // TDM issue for one 128x32 f32 B tile (wave-level; EXEC ignored)
    auto issue_tdm = [&](int k0, unsigned int lds_addr) {
        const unsigned long long ga =
            (unsigned long long)(uintptr_t)(Wt + (size_t)bn * KD + k0);
        v4u g0;
        g0[0] = 1u;                                   // count=1 (valid D#)
        g0[1] = lds_addr;                             // lds_addr
        g0[2] = (unsigned int)(ga & 0xFFFFFFFFu);     // global_addr[31:0]
        g0[3] = (unsigned int)((ga >> 32) & 0x01FFFFFFu) | (2u << 30); // [56:32]|type=2

        v8i32 g1;
        // data_size=4B(2), pad_enable, pad_interval=32dw(4), pad_amount=4dw(3)
        g1[0] = (2 << 16) | (1 << 20) | (4 << 22) | (3 << 25);
        g1[1] = (int)((512u & 0xFFFFu) << 16);        // tensor_dim0 = 512 (lo16)
        g1[2] = (int)(((512u >> 16) & 0xFFFFu) | ((rows_rem & 0xFFFFu) << 16));
        g1[3] = (int)(((rows_rem >> 16) & 0xFFFFu) | (32u << 16)); // tile_dim0=32
        g1[4] = 128;                                  // tile_dim1=128, tile_dim2=0
        g1[5] = 512;                                  // tensor_dim0_stride lo32
        g1[6] = 0;
        g1[7] = 0;

        const v4i32 gz4 = {0, 0, 0, 0};
        const v8i32 gz8 = {0, 0, 0, 0, 0, 0, 0, 0};
        __builtin_amdgcn_tensor_load_to_lds(g0, g1, gz4, gz4, gz8, 0);
    };

    // Async A tile: 4 x b128 per thread (= 4 wave-level async ops)
    auto issue_a = [&](int k0, float (*buf)[FPITCH]) {
        #pragma unroll
        for (int it = 0; it < 4; ++it) {
            const int v   = tid + (it << 8);   // 0..1023
            const int row = v >> 3;            // 8 x float4 per 32-f32 row
            const int c4  = (v & 7) << 2;
            const float* gsrc = A + (size_t)(bm + row) * KD + k0 + c4;
            const unsigned int ldst = (unsigned int)(uintptr_t)&buf[row][c4];
            __builtin_amdgcn_global_load_async_to_lds_b128(
                (gv4i_t*)(uintptr_t)gsrc, (lv4i_t*)ldst, 0, 0);
        }
    };

    // prologue: stage chunk 0
    issue_a(0, Af[0]);
    if (w == 0) issue_tdm(0, ldsB[0]);

    for (int kc = 0; kc < KD / 32; ++kc) {
        const int cur = kc & 1;
        const int nxt = cur ^ 1;
        __syncthreads();   // everyone done computing on buffers [nxt]

        if (kc + 1 < KD / 32) {
            issue_a((kc + 1) << 5, Af[nxt]);                  // prefetch A k+1
            if (w == 0) issue_tdm((kc + 1) << 5, ldsB[nxt]);  // prefetch B k+1
            __builtin_amdgcn_s_wait_asynccnt(4);              // A_k drained
            if (w == 0) __builtin_amdgcn_s_wait_tensorcnt(1); // B_k drained
        } else {
            __builtin_amdgcn_s_wait_asynccnt(0);
            if (w == 0) __builtin_amdgcn_s_wait_tensorcnt(0);
        }
        __syncthreads();   // current tiles visible to all waves

        v16bf af[2], bf[4];
        #pragma unroll
        for (int i = 0; i < 2; ++i)
            af[i] = load_a_frag(Af[cur], m_wave + (i << 4), lane);
        #pragma unroll
        for (int j = 0; j < 4; ++j)
            bf[j] = load_b_frag(Bf[cur], n_wave + (j << 4), lane);

        #pragma unroll
        for (int i = 0; i < 2; ++i)
            #pragma unroll
            for (int j = 0; j < 4; ++j)
                acc[i][j] = __builtin_amdgcn_wmma_f32_16x16x32_bf16(
                    false, af[i], false, bf[j],
                    (short)0, acc[i][j], false, false);
    }

    // epilogue: C layout — VGPR r: lanes 0-15 -> M=r, lanes 16-31 -> M=r+8
    const int laneN = lane & 15;
    const int mOff  = (lane >> 4) << 3;
    #pragma unroll
    for (int i = 0; i < 2; ++i) {
        #pragma unroll
        for (int j = 0; j < 4; ++j) {
            const int gn = bn + n_wave + (j << 4) + laneN;
            if (gn < N) {
                const float bb = bias[gn];
                const int gmBase = bm + m_wave + (i << 4) + mOff;
                #pragma unroll
                for (int r = 0; r < 8; ++r)
                    out[(size_t)(gmBase + r) * N + gn] = acc[i][j][r] + bb;
            }
        }
    }
}

// ---------------------------------------------------------------------------
// In-place row-wise log_softmax. blockIdx.x < T -> globals row, else senses.
// ---------------------------------------------------------------------------
#define SM_THREADS 512

__device__ __forceinline__ float block_reduce_max(float v, float* sm) {
    #pragma unroll
    for (int off = 16; off; off >>= 1) v = fmaxf(v, __shfl_xor(v, off, 32));
    const int wid = threadIdx.x >> 5, lane = threadIdx.x & 31;
    if (lane == 0) sm[wid] = v;
    __syncthreads();
    if (wid == 0) {
        v = (lane < (SM_THREADS >> 5)) ? sm[lane] : -INFINITY;
        #pragma unroll
        for (int off = 8; off; off >>= 1) v = fmaxf(v, __shfl_xor(v, off, 32));
        if (lane == 0) sm[0] = v;
    }
    __syncthreads();
    v = sm[0];
    __syncthreads();
    return v;
}

__device__ __forceinline__ float block_reduce_sum(float v, float* sm) {
    #pragma unroll
    for (int off = 16; off; off >>= 1) v += __shfl_xor(v, off, 32);
    const int wid = threadIdx.x >> 5, lane = threadIdx.x & 31;
    if (lane == 0) sm[wid] = v;
    __syncthreads();
    if (wid == 0) {
        v = (lane < (SM_THREADS >> 5)) ? sm[lane] : 0.0f;
        #pragma unroll
        for (int off = 8; off; off >>= 1) v += __shfl_xor(v, off, 32);
        if (lane == 0) sm[0] = v;
    }
    __syncthreads();
    v = sm[0];
    __syncthreads();
    return v;
}

__global__ void __launch_bounds__(SM_THREADS)
logsoftmax_kernel(float* __restrict__ out) {
    __shared__ float sm[SM_THREADS >> 5];
    const int row = blockIdx.x;
    float* base;
    int N;
    if (row < T_STEPS) {
        base = out + (size_t)row * NG;
        N = NG;
    } else {
        base = out + (size_t)T_STEPS * NG + (size_t)(row - T_STEPS) * NS;
        N = NS;
    }
    float m = -INFINITY;
    for (int i = threadIdx.x; i < N; i += SM_THREADS) m = fmaxf(m, base[i]);
    m = block_reduce_max(m, sm);

    float s = 0.0f;
    for (int i = threadIdx.x; i < N; i += SM_THREADS) s += __expf(base[i] - m);
    s = block_reduce_sum(s, sm);

    const float lse = m + __logf(s);
    for (int i = threadIdx.x; i < N; i += SM_THREADS) base[i] = base[i] - lse;
}

// ---------------------------------------------------------------------------
// Host launcher
//   d_in: 0=X 1=W1 2=b1 3=W2 4=b2 5=Wg 6=bg 7=Ws 8=bs 9=indices
//   d_out: [T,NG] log_softmax globals then [T,NS] log_softmax senses
//   d_ws: h2s (T*H2 f32) | h1 ping-pong (2*H1) | zeros (H2) | barrier ints
// ---------------------------------------------------------------------------
extern "C" void kernel_launch(void* const* d_in, const int* in_sizes, int n_in,
                              void* d_out, int out_size, void* d_ws, size_t ws_size,
                              hipStream_t stream) {
    const float* X   = (const float*)d_in[0];
    const float* W1  = (const float*)d_in[1];
    const float* b1  = (const float*)d_in[2];
    const float* W2  = (const float*)d_in[3];
    const float* b2  = (const float*)d_in[4];
    const float* Wg  = (const float*)d_in[5];
    const float* bg  = (const float*)d_in[6];
    const float* Ws  = (const float*)d_in[7];
    const float* bs  = (const float*)d_in[8];
    const int*   idx = (const int*)d_in[9];
    float* out = (float*)d_out;

    float* h2s   = (float*)d_ws;                     // T*H2
    float* h1buf = h2s + (size_t)T_STEPS * H2DIM;    // 2*H1
    float* zbuf  = h1buf + 2 * H1;                   // H2
    int*   bar   = (int*)(zbuf + H2DIM);             // 4 ints

    init_ws_kernel<<<(2 * H1 + 255) / 256, 256, 0, stream>>>(h1buf, zbuf, bar);

    rnn_kernel<<<RNN_BLOCKS, RNN_THREADS, 0, stream>>>(
        X, W1, b1, W2, b2, idx, h2s, h1buf, zbuf, bar);

    dim3 blk(256);
    dim3 grd_g((NG + 127) / 128, T_STEPS / 128);
    gemm_bias_kernel<<<grd_g, blk, 0, stream>>>(h2s, Wg, bg, out, NG);
    dim3 grd_s((NS + 127) / 128, T_STEPS / 128);
    gemm_bias_kernel<<<grd_s, blk, 0, stream>>>(
        h2s, Ws, bs, out + (size_t)T_STEPS * NG, NS);

    logsoftmax_kernel<<<2 * T_STEPS, SM_THREADS, 0, stream>>>(out);
}
